// MemNet_71786083385515
// MI455X (gfx1250) — matrix operand order repaired and checked by reading, not compiled, not run on hardware
//
#include <hip/hip_runtime.h>
#include <float.h>

// ---------------------------------------------------------------------------
// MemNet on MI455X (gfx1250, wave32).
// Conv1d(k=5, 256->256) dominates: 5 GEMMs of (B*S)x256x256 per hop.
// bf16 WMMA 16x16x32 with hi/lo error-compensated split (3 wmma/tile-step,
// ~fp32 accuracy), f32 accumulate. xn is stored as planar bf16 hi/lo so the
// A-tile staging for hops>=1 is a raw byte copy -> global_load_async_to_lds
// (ASYNCcnt path), plus global_prefetch for the streamed B weights.
// ---------------------------------------------------------------------------

#define B_     32
#define S_     4096
#define D_     256
#define TAPS   5
#define MAXHOP 10

#define TM    128          // block output rows (s positions)
#define ROWS  (TM + 4)     // staged rows incl. conv halo (132)
#define LSTR  264          // LDS row stride in bf16 elems (256 + 8 pad)

typedef __attribute__((ext_vector_type(16))) __bf16 v16bf;
typedef __attribute__((ext_vector_type(8)))  __bf16 v8bf;
typedef __attribute__((ext_vector_type(8)))  float  v8f;
typedef int v4i __attribute__((__vector_size__(16)));

__device__ __forceinline__ unsigned short f2bf(float f) {
    unsigned u = __float_as_uint(f);
    u += 0x7FFFu + ((u >> 16) & 1u);           // round-to-nearest-even
    return (unsigned short)(u >> 16);
}
__device__ __forceinline__ float bf2f(unsigned short h) {
    return __uint_as_float(((unsigned)h) << 16);
}

// ---- CDNA5 async global->LDS copy (ASYNCcnt) ------------------------------
__device__ __forceinline__ void async_copy16(unsigned short* ldsDst,
                                             const unsigned short* gSrc) {
#if __has_builtin(__builtin_amdgcn_global_load_async_to_lds_b128)
    __builtin_amdgcn_global_load_async_to_lds_b128(
        (__attribute__((address_space(1))) v4i*)gSrc,
        (__attribute__((address_space(3))) v4i*)ldsDst,
        0, 0);
#else
    unsigned ldsOff = (unsigned)(size_t)ldsDst;   // LDS aperture: addr[31:0]
    asm volatile("global_load_async_to_lds_b128 %0, %1, off"
                 :: "v"(ldsOff), "v"(gSrc) : "memory");
#endif
}
__device__ __forceinline__ void wait_async0() {
#if __has_builtin(__builtin_amdgcn_s_wait_asynccnt)
    __builtin_amdgcn_s_wait_asynccnt(0);
#else
    asm volatile("s_wait_asynccnt 0x0" ::: "memory");
#endif
}

// ---------------------------------------------------------------------------
// Pack weights to bf16 hi/lo in [tap][n][k] layout; fold BN into scale/bias;
// init q = broadcast(query); runFlag = 1.
// ---------------------------------------------------------------------------
__global__ void pack_kernel(const float* __restrict__ conv_w,
                            const float* __restrict__ conv_b,
                            const float* __restrict__ gamma,
                            const float* __restrict__ beta,
                            const float* __restrict__ mean,
                            const float* __restrict__ var,
                            const float* __restrict__ query,
                            unsigned short* __restrict__ Whi,
                            unsigned short* __restrict__ Wlo,
                            float* __restrict__ scale,
                            float* __restrict__ bias2,
                            float* __restrict__ q,
                            int* __restrict__ runFlag)
{
    int idx = blockIdx.x * 256 + threadIdx.x;     // 1280 blocks -> 327680
    if (idx < TAPS * D_ * D_) {
        int t   = idx / (D_ * D_);
        int rem = idx % (D_ * D_);
        int n   = rem / D_;
        int k   = rem % D_;
        float w = conv_w[(n * D_ + k) * TAPS + t];   // conv_w[O][I][H]
        unsigned short hi = f2bf(w);
        float lo = w - bf2f(hi);
        Whi[(t * D_ + n) * D_ + k] = hi;
        Wlo[(t * D_ + n) * D_ + k] = f2bf(lo);
    }
    if (idx < D_) {
        float sc = gamma[idx] * rsqrtf(var[idx] + 1e-5f);
        scale[idx] = sc;
        bias2[idx] = beta[idx] + (conv_b[idx] - mean[idx]) * sc;
    }
    if (idx < B_ * D_) q[idx] = query[idx % D_];
    if (idx == 0) *runFlag = 1;
}

// ---------------------------------------------------------------------------
// e[b,s] = mask ? -FLT_MAX : dot(xc[b,s,:], q[b,:]).  One wave per row.
// f32 input variant (hop 0).
// ---------------------------------------------------------------------------
__global__ void dot_kernel(const float* __restrict__ x,
                           const float* __restrict__ q,
                           const unsigned char* __restrict__ mask,
                           float* __restrict__ e)
{
    int wave = threadIdx.x >> 5;
    int lane = threadIdx.x & 31;
    int row  = blockIdx.x * 8 + wave;             // over B*S
    int b    = row >> 12;                         // /4096
    const float4* xr = (const float4*)(x + (size_t)row * D_);
    const float4* qr = (const float4*)(q + b * D_);
    float acc = 0.f;
#pragma unroll
    for (int i = 0; i < 2; ++i) {
        float4 xv = xr[lane * 2 + i];
        float4 qv = qr[lane * 2 + i];
        acc += xv.x * qv.x + xv.y * qv.y + xv.z * qv.z + xv.w * qv.w;
    }
#pragma unroll
    for (int off = 16; off; off >>= 1) acc += __shfl_xor(acc, off, 32);
    if (lane == 0) e[row] = mask[row] ? -FLT_MAX : acc;
}

// Split bf16 hi/lo input variant (hops >= 1).
__global__ void dot_split_kernel(const unsigned short* __restrict__ xh,
                                 const unsigned short* __restrict__ xl,
                                 const float* __restrict__ q,
                                 const unsigned char* __restrict__ mask,
                                 float* __restrict__ e)
{
    int wave = threadIdx.x >> 5;
    int lane = threadIdx.x & 31;
    int row  = blockIdx.x * 8 + wave;
    int b    = row >> 12;
    uint4 hv = *(const uint4*)(xh + (size_t)row * D_ + lane * 8);
    uint4 lv = *(const uint4*)(xl + (size_t)row * D_ + lane * 8);
    const float* qr = q + b * D_ + lane * 8;
    float acc = 0.f;
    const unsigned* hw = (const unsigned*)&hv;
    const unsigned* lw = (const unsigned*)&lv;
#pragma unroll
    for (int j = 0; j < 4; ++j) {
        float f0 = bf2f((unsigned short)hw[j]) + bf2f((unsigned short)lw[j]);
        float f1 = bf2f((unsigned short)(hw[j] >> 16)) + bf2f((unsigned short)(lw[j] >> 16));
        acc += f0 * qr[2 * j] + f1 * qr[2 * j + 1];
    }
#pragma unroll
    for (int off = 16; off; off >>= 1) acc += __shfl_xor(acc, off, 32);
    if (lane == 0) e[row] = mask[row] ? -FLT_MAX : acc;
}

// ---------------------------------------------------------------------------
// softmax over S per batch row.
// ---------------------------------------------------------------------------
__global__ void softmax_kernel(const float* __restrict__ e, float* __restrict__ p)
{
    __shared__ float red[256];
    int b = blockIdx.x;
    const float* er = e + (size_t)b * S_;
    float* pr = p + (size_t)b * S_;

    float m = -FLT_MAX;
    for (int s = threadIdx.x; s < S_; s += 256) m = fmaxf(m, er[s]);
    red[threadIdx.x] = m; __syncthreads();
    for (int off = 128; off; off >>= 1) {
        if (threadIdx.x < off) red[threadIdx.x] = fmaxf(red[threadIdx.x], red[threadIdx.x + off]);
        __syncthreads();
    }
    m = red[0]; __syncthreads();

    float sum = 0.f;
    for (int s = threadIdx.x; s < S_; s += 256) {
        float v = expf(er[s] - m);
        pr[s] = v;
        sum += v;
    }
    red[threadIdx.x] = sum; __syncthreads();
    for (int off = 128; off; off >>= 1) {
        if (threadIdx.x < off) red[threadIdx.x] += red[threadIdx.x + off];
        __syncthreads();
    }
    float inv = 1.f / red[0];
    for (int s = threadIdx.x; s < S_; s += 256) pr[s] *= inv;
}

// ---------------------------------------------------------------------------
// Conv1d(k=5) + BN + leaky-relu + mask, as a WMMA GEMM.
// Block: 128 rows x 256 cols. 8 waves in a 2x4 grid, each 64x64 (16 C tiles).
// splitInput=0: xinF (f32), convert to hi/lo in registers.
// splitInput=1: xinHi/xinLo (bf16 planes), async global->LDS byte copy.
// Output always planar bf16 hi/lo.
// ---------------------------------------------------------------------------
__global__ __launch_bounds__(256) void conv_kernel(
    const float* __restrict__ xinF,
    const unsigned short* __restrict__ xinHi,
    const unsigned short* __restrict__ xinLo,
    const unsigned short* __restrict__ Whi,
    const unsigned short* __restrict__ Wlo,
    const float* __restrict__ scale,
    const float* __restrict__ bias2,
    const unsigned char* __restrict__ mask,
    unsigned short* __restrict__ xoutHi,
    unsigned short* __restrict__ xoutLo,
    int splitInput)
{
    extern __shared__ unsigned short lds[];       // Ah[ROWS*LSTR] | Al[ROWS*LSTR]
    unsigned short* Ah = lds;
    unsigned short* Al = lds + ROWS * LSTR;

    int b    = blockIdx.x >> 5;                   // 32 s-tiles per batch
    int tile = blockIdx.x & 31;
    int s0   = tile * TM;

    if (splitInput) {
        // ---- Async byte-copy staging: 16B chunks per lane, per plane ----
        for (int i = threadIdx.x; i < ROWS * 32; i += 256) {
            int r = i >> 5;                       // staged row 0..131
            int c = i & 31;                       // 16B chunk (8 bf16)
            int s = s0 - 2 + r;
            unsigned short* dh = Ah + r * LSTR + c * 8;
            unsigned short* dl = Al + r * LSTR + c * 8;
            if (s >= 0 && s < S_) {
                size_t g = ((size_t)(b * S_ + s)) * D_ + c * 8;
                async_copy16(dh, xinHi + g);
                async_copy16(dl, xinLo + g);
            } else {                              // batch-edge halo: zero fill
                uint4 z = make_uint4(0u, 0u, 0u, 0u);
                *(uint4*)dh = z;
                *(uint4*)dl = z;
            }
        }
        wait_async0();
        __syncthreads();
    } else {
        // ---- f32 input: convert to hi/lo bf16 while staging ----
        for (int i = threadIdx.x; i < ROWS * 64; i += 256) {
            int r  = i >> 6;                      // staged row 0..131
            int cv = i & 63;                      // float4 column
            int s  = s0 - 2 + r;
            float4 v = make_float4(0.f, 0.f, 0.f, 0.f);
            if (s >= 0 && s < S_)
                v = *(const float4*)(xinF + ((size_t)(b * S_ + s)) * D_ + cv * 4);
            unsigned short h0 = f2bf(v.x), h1 = f2bf(v.y), h2 = f2bf(v.z), h3 = f2bf(v.w);
            unsigned short l0 = f2bf(v.x - bf2f(h0));
            unsigned short l1 = f2bf(v.y - bf2f(h1));
            unsigned short l2 = f2bf(v.z - bf2f(h2));
            unsigned short l3 = f2bf(v.w - bf2f(h3));
            uint2 hp, lp;
            hp.x = (unsigned)h0 | ((unsigned)h1 << 16);
            hp.y = (unsigned)h2 | ((unsigned)h3 << 16);
            lp.x = (unsigned)l0 | ((unsigned)l1 << 16);
            lp.y = (unsigned)l2 | ((unsigned)l3 << 16);
            *(uint2*)(Ah + r * LSTR + cv * 4) = hp;
            *(uint2*)(Al + r * LSTR + cv * 4) = lp;
        }
        __syncthreads();
    }

    int lane     = threadIdx.x & 31;
    int wave     = threadIdx.x >> 5;
    int wm       = wave >> 2;                     // 0..1 (row half)
    int wn       = wave & 3;                      // 0..3 (col quarter)
    int laneHigh = lane >> 4;                     // 0/1
    int lane16   = lane & 15;

    v8f acc[4][4];
#pragma unroll
    for (int mi = 0; mi < 4; ++mi)
#pragma unroll
        for (int ni = 0; ni < 4; ++ni)
            acc[mi][ni] = (v8f){0.f, 0.f, 0.f, 0.f, 0.f, 0.f, 0.f, 0.f};

    const int rowBase = wm * 64;

#pragma unroll 1
    for (int t = 0; t < TAPS; ++t) {
        const unsigned short* Wh = Whi + (size_t)t * D_ * D_;
        const unsigned short* Wl = Wlo + (size_t)t * D_ * D_;
#pragma unroll 1
        for (int kk = 0; kk < D_; kk += 32) {
            // Prefetch next k-step's weight rows into the cache hierarchy.
            {
                int kn = (kk + 32) & 255;
                const unsigned short* pf = Wh + (size_t)(wn * 64 + lane16) * D_ + kn;
                __builtin_prefetch(pf, 0, 0);
                const unsigned short* pfl = Wl + (size_t)(wn * 64 + lane16) * D_ + kn;
                __builtin_prefetch(pfl, 0, 0);
            }
            // A fragments: ISA 16-bit A 16x32 layout — lane L holds row
            // m=L%16, two contiguous 8-chunks at k0 and k0+16, k0 = kk + 8*(L>=16).
            int k0 = kk + laneHigh * 8;
            v16bf ah[4], al[4];
#pragma unroll
            for (int mi = 0; mi < 4; ++mi) {
                int r = rowBase + mi * 16 + lane16 + t;   // staged row = out_row + tap
                const unsigned short* ph = Ah + r * LSTR + k0;
                const unsigned short* pl = Al + r * LSTR + k0;
                union { v16bf v; v8bf h[2]; } ua, ul;
                ua.h[0] = *(const v8bf*)(ph);
                ua.h[1] = *(const v8bf*)(ph + 16);
                ul.h[0] = *(const v8bf*)(pl);
                ul.h[1] = *(const v8bf*)(pl + 16);
                ah[mi] = ua.v;
                al[mi] = ul.v;
            }
#pragma unroll
            for (int ni = 0; ni < 4; ++ni) {
                // B fragment: lane L holds col n=L%16, two contiguous 8-chunks at
                // kb and kb+16, kb = kk + 8*(L>=16). Weights stored [t][n][k].
                int n  = wn * 64 + ni * 16 + lane16;
                int kb = kk + laneHigh * 8;
                union { v16bf v; v8bf h[2]; } ubh, ubl;
                ubh.h[0] = *(const v8bf*)(Wh + (size_t)n * D_ + kb);
                ubh.h[1] = *(const v8bf*)(Wh + (size_t)n * D_ + kb + 16);
                ubl.h[0] = *(const v8bf*)(Wl + (size_t)n * D_ + kb);
                ubl.h[1] = *(const v8bf*)(Wl + (size_t)n * D_ + kb + 16);
                v16bf bh = ubh.v;
                v16bf bl = ubl.v;
#pragma unroll
                for (int mi = 0; mi < 4; ++mi) {
                    acc[mi][ni] = __builtin_amdgcn_wmma_f32_16x16x32_bf16(
                        false, ah[mi], false, bh, (short)0, acc[mi][ni], false, false);
                    acc[mi][ni] = __builtin_amdgcn_wmma_f32_16x16x32_bf16(
                        false, ah[mi], false, bl, (short)0, acc[mi][ni], false, false);
                    acc[mi][ni] = __builtin_amdgcn_wmma_f32_16x16x32_bf16(
                        false, al[mi], false, bh, (short)0, acc[mi][ni], false, false);
                }
            }
        }
    }

    // ---- Epilogue: affine(BN) + leaky relu + mask zero, planar hi/lo store ----
#pragma unroll
    for (int mi = 0; mi < 4; ++mi) {
#pragma unroll
        for (int ni = 0; ni < 4; ++ni) {
            int n = wn * 64 + ni * 16 + lane16;
            float sc = scale[n];
            float bs = bias2[n];
#pragma unroll
            for (int r = 0; r < 8; ++r) {
                int lr = wm * 64 + mi * 16 + laneHigh * 8 + r;  // C layout: M = r + 8*(lane>=16)
                int s  = s0 + lr;
                float v = acc[mi][ni][r] * sc + bs;
                v = (v >= 0.f) ? v : 0.01f * v;
                if (mask[b * S_ + s]) v = 0.f;
                unsigned short h = f2bf(v);
                unsigned short l = f2bf(v - bf2f(h));
                size_t o = ((size_t)(b * S_ + s)) * D_ + n;
                xoutHi[o] = h;
                xoutLo[o] = l;
            }
        }
    }
}

// ---------------------------------------------------------------------------
// Partial o[b,d] = sum over an S-chunk of xn[b,s,d] * probs[b,s].
// ---------------------------------------------------------------------------
__global__ void opart_kernel(const unsigned short* __restrict__ xh,
                             const unsigned short* __restrict__ xl,
                             const float* __restrict__ p,
                             float* __restrict__ opart)
{
    int b  = blockIdx.x & 31;
    int sc = blockIdx.x >> 5;                     // 8 chunks of 512
    int d  = threadIdx.x;
    const unsigned short* xhb = xh + ((size_t)b * S_ + sc * 512) * D_;
    const unsigned short* xlb = xl + ((size_t)b * S_ + sc * 512) * D_;
    const float* pb = p + (size_t)b * S_ + sc * 512;
    float acc = 0.f;
    for (int s = 0; s < 512; ++s) {
        float v = bf2f(xhb[(size_t)s * D_ + d]) + bf2f(xlb[(size_t)s * D_ + d]);
        acc = fmaf(v, pb[s], acc);
    }
    opart[(sc * B_ + b) * D_ + d] = acc;
}

// ---------------------------------------------------------------------------
// Finalize one hop: if running, q += o; halt if argmax(halt logits)==1.
// (while running, all halted[b]==false, so q_new = o + q exactly.)
// ---------------------------------------------------------------------------
__global__ void finalize_kernel(const float* __restrict__ opart,
                                float* __restrict__ q,
                                const float* __restrict__ halt_w,
                                const float* __restrict__ halt_b,
                                int* __restrict__ runFlag)
{
    __shared__ float red0[256];
    __shared__ float red1[256];
    __shared__ int haltAny;
    if (*runFlag == 0) return;                    // uniform across block
    if (threadIdx.x == 0) haltAny = 0;
    __syncthreads();

    int d = threadIdx.x;
    float w0 = halt_w[d];
    float w1 = halt_w[D_ + d];
    for (int b = 0; b < B_; ++b) {
        float o = 0.f;
#pragma unroll
        for (int sc = 0; sc < 8; ++sc) o += opart[(sc * B_ + b) * D_ + d];
        float qn = o + q[b * D_ + d];
        q[b * D_ + d] = qn;
        red0[d] = qn * w0;
        red1[d] = qn * w1;
        __syncthreads();
        for (int off = 128; off; off >>= 1) {
            if (d < off) { red0[d] += red0[d + off]; red1[d] += red1[d + off]; }
            __syncthreads();
        }
        if (d == 0) {
            float l0 = red0[0] + halt_b[0];
            float l1 = red1[0] + halt_b[1];
            if (l1 > l0) haltAny = 1;
        }
        __syncthreads();
    }
    if (threadIdx.x == 0 && haltAny) *runFlag = 0;
}

// ---------------------------------------------------------------------------
extern "C" void kernel_launch(void* const* d_in, const int* in_sizes, int n_in,
                              void* d_out, int out_size, void* d_ws, size_t ws_size,
                              hipStream_t stream)
{
    const float*         x      = (const float*)d_in[0];
    const unsigned char* mask   = (const unsigned char*)d_in[1];  // jax bool_ = 1 byte
    const float*         query  = (const float*)d_in[2];
    const float*         conv_w = (const float*)d_in[3];
    const float*         conv_b = (const float*)d_in[4];
    const float*         gamma  = (const float*)d_in[5];
    const float*         beta   = (const float*)d_in[6];
    const float*         mean   = (const float*)d_in[7];
    const float*         var    = (const float*)d_in[8];
    const float*         halt_w = (const float*)d_in[9];
    const float*         halt_b = (const float*)d_in[10];

    char* ws = (char*)d_ws;
    size_t off = 0;
    auto alloc = [&](size_t bytes) -> void* {
        void* p = ws + off;
        off += (bytes + 255) & ~(size_t)255;
        return p;
    };
    unsigned short* xh0     = (unsigned short*)alloc((size_t)B_ * S_ * D_ * 2);
    unsigned short* xl0     = (unsigned short*)alloc((size_t)B_ * S_ * D_ * 2);
    unsigned short* xh1     = (unsigned short*)alloc((size_t)B_ * S_ * D_ * 2);
    unsigned short* xl1     = (unsigned short*)alloc((size_t)B_ * S_ * D_ * 2);
    float*          e       = (float*)alloc((size_t)B_ * S_ * 4);
    float*          probs   = (float*)alloc((size_t)B_ * S_ * 4);
    float*          q       = (float*)alloc((size_t)B_ * D_ * 4);
    float*          opart   = (float*)alloc((size_t)8 * B_ * D_ * 4);
    unsigned short* Whi     = (unsigned short*)alloc((size_t)TAPS * D_ * D_ * 2);
    unsigned short* Wlo     = (unsigned short*)alloc((size_t)TAPS * D_ * D_ * 2);
    float*          scale   = (float*)alloc(D_ * 4);
    float*          bias2   = (float*)alloc(D_ * 4);
    int*            runFlag = (int*)alloc(256);

    pack_kernel<<<1280, 256, 0, stream>>>(conv_w, conv_b, gamma, beta, mean, var,
                                          query, Whi, Wlo, scale, bias2, q, runFlag);

    size_t ldsBytes = (size_t)2 * ROWS * LSTR * sizeof(unsigned short); // ~139 KB (<320 KB/WGP)

    for (int h = 0; h < MAXHOP; ++h) {
        const unsigned short* xhIn = (h & 1) ? xh0 : xh1;   // valid for h>=1
        const unsigned short* xlIn = (h & 1) ? xl0 : xl1;
        unsigned short*       xhOut = (h & 1) ? xh1 : xh0;
        unsigned short*       xlOut = (h & 1) ? xl1 : xl0;

        if (h == 0)
            dot_kernel<<<B_ * S_ / 8, 256, 0, stream>>>(x, q, mask, e);
        else
            dot_split_kernel<<<B_ * S_ / 8, 256, 0, stream>>>(xhIn, xlIn, q, mask, e);

        softmax_kernel<<<B_, 256, 0, stream>>>(e, probs);

        conv_kernel<<<B_ * (S_ / TM), 256, ldsBytes, stream>>>(
            (h == 0) ? x : (const float*)nullptr, xhIn, xlIn,
            Whi, Wlo, scale, bias2, mask, xhOut, xlOut, (h == 0) ? 0 : 1);

        opart_kernel<<<B_ * 8, 256, 0, stream>>>(xhOut, xlOut, probs, opart);
        finalize_kernel<<<1, 256, 0, stream>>>(opart, q, halt_w, halt_b, runFlag);
    }

    (void)hipMemcpyAsync(d_out, q, (size_t)B_ * D_ * 4, hipMemcpyDeviceToDevice, stream);
    (void)in_sizes; (void)n_in; (void)out_size; (void)ws_size;
}
